// Conv2dLayer_3195455668909
// MI455X (gfx1250) — matrix operand order repaired and checked by reading, not compiled, not run on hardware
//
#include <hip/hip_runtime.h>

// ---------------------------------------------------------------------------
// Types / helpers
// ---------------------------------------------------------------------------
typedef __attribute__((ext_vector_type(16))) __bf16 v16bf;
typedef __attribute__((ext_vector_type(8)))  float  v8f;

struct alignas(16) U4 { unsigned x, y, z, w; };
struct U8 { U4 lo, hi; };

__device__ __forceinline__ v16bf ld_bf16_2x8(const unsigned short* p0,
                                             const unsigned short* p1) {
  U8 t;
  t.lo = *(const U4*)(const void*)p0;   // 16B aligned LDS read -> ds_load_b128
  t.hi = *(const U4*)(const void*)p1;
  return __builtin_bit_cast(v16bf, t);
}

__device__ __forceinline__ unsigned short f2bf(float f) {  // RNE f32->bf16
  unsigned u = __builtin_bit_cast(unsigned, f);
  u += 0x7FFFu + ((u >> 16) & 1u);
  return (unsigned short)(u >> 16);
}

// Async global->LDS copy (GVS mode: 64-bit SGPR base + 32-bit VGPR offset).
__device__ __forceinline__ void async_load_b128(unsigned lds_off, unsigned goff,
                                                const void* base) {
  asm volatile("global_load_async_to_lds_b128 %0, %1, %2"
               :: "v"(lds_off), "v"(goff), "s"(base) : "memory");
}
__device__ __forceinline__ void wait_async0() {
#if __has_builtin(__builtin_amdgcn_s_wait_asynccnt)
  __builtin_amdgcn_s_wait_asynccnt(0);
#else
  asm volatile("s_wait_asynccnt 0" ::: "memory");
#endif
}

#define WMMA_BF16(acc, A, B)                                                   \
  acc = __builtin_amdgcn_wmma_f32_16x16x32_bf16(false, (A), false, (B),        \
                                                (short)0, (acc), false, false)

// ---------------------------------------------------------------------------
// Problem constants
// ---------------------------------------------------------------------------
constexpr int NB = 8, CIN = 512, OC = 256, HW = 64, OHW = 128;
constexpr int KCH = 32;            // input channels per WMMA K-chunk
constexpr int NCHUNK = CIN / KCH;  // 16
constexpr int TA = 6;              // base rows per block  -> 12 out rows
constexpr int TB = 14;             // base cols per block  -> 28 out cols
constexpr float WGAIN = 0.014731391f;   // 1/sqrt(512*3*3)

union SM {
  struct Stage {
    unsigned short xs[2][10][18][40];   // [buf][row][col][cin]   (bf16 bits)
    unsigned short wl[2][9][16][32];    // [buf][tap][m][cin]
  } s;
  float P[4][8][16][16];                // [phase][waverow][m][n]
};

// ---------------------------------------------------------------------------
// Prep kernel 1: x f32 NCHW -> bf16 NHWC (tiled transpose, coalesced both ways)
// ---------------------------------------------------------------------------
__global__ __launch_bounds__(256) void cvt_x(const float* __restrict__ x,
                                             unsigned short* __restrict__ xb) {
  __shared__ float t[32][33];
  const int hw0 = blockIdx.x * 32, c0 = blockIdx.y * 32, n = blockIdx.z;
  const int tx = threadIdx.x, ty = threadIdx.y;
#pragma unroll
  for (int i = 0; i < 4; ++i) {
    int c = c0 + ty + i * 8, hw = hw0 + tx;
    t[ty + i * 8][tx] = x[((size_t)n * CIN + c) * (HW * HW) + hw];
  }
  __syncthreads();
#pragma unroll
  for (int i = 0; i < 4; ++i) {
    int hw = hw0 + ty + i * 8, c = c0 + tx;
    xb[((size_t)n * (HW * HW) + hw) * CIN + c] = f2bf(t[tx][ty + i * 8]);
  }
}

// ---------------------------------------------------------------------------
// Prep kernel 2: w f32 [oc][cin][3][3] * gain -> bf16 [oc][tap][cin]
// ---------------------------------------------------------------------------
__global__ __launch_bounds__(256) void cvt_w(const float* __restrict__ w,
                                             unsigned short* __restrict__ wb) {
  int e = blockIdx.x * 256 + threadIdx.x;           // OC*9*CIN = 1179648
  int oc = e / (9 * CIN);
  int r  = e - oc * (9 * CIN);
  int tap = r / CIN;
  int cin = r - tap * CIN;
  wb[e] = f2bf(w[((size_t)oc * CIN + cin) * 9 + tap] * WGAIN);
}

// ---------------------------------------------------------------------------
// Async stage of one K-chunk (x tile + weight tile) into LDS buffer `buf`
// ---------------------------------------------------------------------------
__device__ __forceinline__ void issue_chunk(SM& sm, int tid, int n, int oc0,
                                            int a0, int b0, int kc, int buf,
                                            const unsigned short* xb,
                                            const unsigned short* wb) {
  // x: 9 rows x 17 cols x 4 b128-pieces = 612 transfers
#pragma unroll
  for (int t = 0; t < 3; ++t) {
    int idx = t * 256 + tid;
    if (idx < 612) {
      int q = idx & 3, cell = idx >> 2;
      int lc = cell % 17, lr = cell / 17;
      int xr = a0 - 2 + lr, xc = b0 - 2 + lc;
      if (xr >= 0 && xr < HW && xc >= 0 && xc < HW) {
        unsigned goff = (unsigned)((((n * HW + xr) * HW + xc) * CIN +
                                    kc * KCH + q * 8) * 2);
        unsigned laddr = (unsigned)(size_t)&sm.s.xs[buf][lr][lc][q * 8];
        async_load_b128(laddr, goff, xb);
      }
    }
  }
  // w: 9 taps x 16 m x 4 pieces = 576 transfers
#pragma unroll
  for (int t = 0; t < 3; ++t) {
    int idx = t * 256 + tid;
    if (idx < 576) {
      int q = idx & 3, cell = idx >> 2;
      int m = cell & 15, tap = cell >> 4;
      unsigned goff = (unsigned)((((oc0 + m) * 9 + tap) * CIN +
                                  kc * KCH + q * 8) * 2);
      unsigned laddr = (unsigned)(size_t)&sm.s.wl[buf][tap][m][q * 8];
      async_load_b128(laddr, goff, wb);
    }
  }
}

// ---------------------------------------------------------------------------
// Fused transposed-conv (phase-decomposed implicit GEMM) + FIR + bias_act
// ---------------------------------------------------------------------------
__global__ __launch_bounds__(256) void convup_fused(
    const unsigned short* __restrict__ xb, const unsigned short* __restrict__ wb,
    const float* __restrict__ bias, float* __restrict__ out) {
  __shared__ alignas(16) SM sm;

  const int tid  = threadIdx.x;
  const int lane = tid & 31, wv = tid >> 5;     // 8 waves = 8 base rows
  const int Nl   = lane & 15, half = lane >> 4; // WMMA lane coords
  const int b0   = blockIdx.x * TB;             // base-grid col origin
  const int a0   = blockIdx.y * TA;             // base-grid row origin
  const int n    = blockIdx.z >> 4;
  const int oc0  = (blockIdx.z & 15) * 16;

  // Zero x tiles once: halo cells are never written by async loads.
  for (int i = tid; i < (int)(sizeof(sm.s.xs) / 4); i += 256)
    ((unsigned*)(void*)sm.s.xs)[i] = 0u;
  __syncthreads();                 // ds stores drained before async writes

  issue_chunk(sm, tid, n, oc0, a0, b0, 0, 0, xb, wb);
  wait_async0();
  __syncthreads();

  v8f p00 = {}, p01 = {}, p10 = {}, p11 = {};   // phases [rowpar][colpar]

  for (int kc = 0; kc < NCHUNK; ++kc) {
    const int cur = kc & 1;
    if (kc + 1 < NCHUNK)
      issue_chunk(sm, tid, n, oc0, a0, b0, kc + 1, (kc + 1) & 1, xb, wb);

    // ---- Batch-load ALL operand tiles first (26 ds_load_b128), then burst
    //      the 9 WMMAs so the matrix pipe isn't gated on per-op dscnt waits.
    // B tiles (32x16 bf16): lanes 0-15 hold K=0..15 (2/VGPR), 16-31 K=16..31.
    // row lr = wv   <-> x row a_w-1 (tap ki=2); lr = wv+1 <-> x row a_w (ki=0,1)
    // col  n+1 <-> taps kj in {0,1};  col n <-> tap kj=2
    const unsigned short(&xsC)[10][18][40] = sm.s.xs[cur];
    const unsigned short* q11 = &xsC[wv + 1][Nl + 1][half * 16];
    const unsigned short* q10 = &xsC[wv + 1][Nl + 0][half * 16];
    const unsigned short* q01 = &xsC[wv + 0][Nl + 1][half * 16];
    const unsigned short* q00 = &xsC[wv + 0][Nl + 0][half * 16];
    v16bf Br1c1 = ld_bf16_2x8(q11, q11 + 8);
    v16bf Br1c0 = ld_bf16_2x8(q10, q10 + 8);
    v16bf Br0c1 = ld_bf16_2x8(q01, q01 + 8);
    v16bf Br0c0 = ld_bf16_2x8(q00, q00 + 8);

    // A tiles (16x32 bf16): lanes 0-15 K 0..15, lanes 16-31 K 16..31
#define ATILE(t)                                                               \
  ld_bf16_2x8(&sm.s.wl[cur][t][Nl][half * 16], &sm.s.wl[cur][t][Nl][half * 16 + 8])
    v16bf A0 = ATILE(0), A1 = ATILE(1), A2 = ATILE(2);
    v16bf A3 = ATILE(3), A4 = ATILE(4), A5 = ATILE(5);
    v16bf A6 = ATILE(6), A7 = ATILE(7), A8 = ATILE(8);
#undef ATILE

    // 9 back-to-back WMMAs, accumulators interleaved.
    WMMA_BF16(p00, A0, Br1c1);   // tap (0,0)
    WMMA_BF16(p01, A1, Br1c1);   // tap (0,1)
    WMMA_BF16(p10, A3, Br1c1);   // tap (1,0)
    WMMA_BF16(p11, A4, Br1c1);   // tap (1,1)
    WMMA_BF16(p00, A2, Br1c0);   // tap (0,2)
    WMMA_BF16(p10, A5, Br1c0);   // tap (1,2)
    WMMA_BF16(p01, A7, Br0c1);   // tap (2,1)
    WMMA_BF16(p00, A6, Br0c1);   // tap (2,0)
    WMMA_BF16(p00, A8, Br0c0);   // tap (2,2)

    wait_async0();
    __syncthreads();
  }

  // Spill phase accumulators to LDS (C layout: VGPR v -> M = v + 8*half, N=lane%16)
#pragma unroll
  for (int v = 0; v < 8; ++v) {
    int M = v + half * 8;
    sm.P[0][wv][M][Nl] = p00[v];
    sm.P[1][wv][M][Nl] = p01[v];
    sm.P[2][wv][M][Nl] = p10[v];
    sm.P[3][wv][M][Nl] = p11[v];
  }
  __syncthreads();

  // 4x4 FIR over phase grids + bias + lrelu*sqrt(2) + clamp
  const float fc[4] = {0.125f, 0.375f, 0.375f, 0.125f};
  const int RPH[2][4] = {{1, 0, 1, 0}, {0, 1, 0, 1}};   // phase parity per tap
  const int RDA[2][4] = {{-1, 0, 0, 1}, {0, 0, 1, 1}};  // base-grid offset

  for (int e = tid; e < 16 * (2 * TA) * (2 * TB); e += 256) {
    int m = e / (2 * TA * 2 * TB);
    int r2 = e - m * (2 * TA * 2 * TB);
    int orow = r2 / (2 * TB), ocol = r2 - orow * (2 * TB);
    int r = 2 * a0 + orow, s = 2 * b0 + ocol;
    if (r >= OHW || s >= OHW) continue;
    int pr = orow & 1, pc = ocol & 1;
    int al = orow >> 1, bl = ocol >> 1;
    float acc = 0.f;
#pragma unroll
    for (int tr = 0; tr < 4; ++tr) {
      int wr = al + 1 + RDA[pr][tr];
      int phr = RPH[pr][tr];
      float fr = fc[tr];
#pragma unroll
      for (int tc = 0; tc < 4; ++tc) {
        int nc = bl + 1 + RDA[pc][tc];
        acc += fr * fc[tc] * sm.P[phr * 2 + RPH[pc][tc]][wr][m][nc];
      }
    }
    float y = acc * 4.0f + bias[oc0 + m];           // upfirdn gain = up^2
    y = (y >= 0.f ? y : 0.2f * y) * 1.41421356237f; // lrelu * sqrt(2)
    y = fminf(fmaxf(y, -256.f), 256.f);
    out[(((size_t)n * OC + oc0 + m) * OHW + r) * OHW + s] = y;
  }
}

// ---------------------------------------------------------------------------
extern "C" void kernel_launch(void* const* d_in, const int* in_sizes, int n_in,
                              void* d_out, int out_size, void* d_ws, size_t ws_size,
                              hipStream_t stream) {
  const float* x = (const float*)d_in[0];
  const float* w = (const float*)d_in[1];
  const float* b = (const float*)d_in[2];
  float* out = (float*)d_out;

  unsigned short* xb = (unsigned short*)d_ws;                 // 33.5 MB bf16 NHWC
  unsigned short* wbq = xb + (size_t)NB * HW * HW * CIN;      // 2.36 MB bf16

  cvt_x<<<dim3((HW * HW) / 32, CIN / 32, NB), dim3(32, 8), 0, stream>>>(x, xb);
  cvt_w<<<(OC * 9 * CIN) / 256, 256, 0, stream>>>(w, wbq);

  dim3 grid((HW + TB - 1) / TB,                 // 5 col tiles
            (HW + TA - 1) / TA,                 // 11 row tiles
            NB * (OC / 16));                    // 128 (n, oc-tile)
  convup_fused<<<grid, 256, 0, stream>>>(xb, wbq, b, out);
}
